// CompoundProteinInteractionPrediction_19112604467865
// MI455X (gfx1250) — compile-verified, matrix operand order checked by reading, not run on hardware
//
#include <hip/hip_runtime.h>
#include <hip/hip_bf16.h>
#include <math.h>

// MI455X / gfx1250, wave32.
// Roofline: adjacency (16384^2 fp32, 0/1, ~0.2% dense) dominates. Sparsify A
// once (1.07GB scan -> ~12MB slab), then each GNN layer is a gather-sum over
// hs (8MB, L2-resident; 192MB L2). WMMA (v_wmma_f32_16x16x32_bf16) drives the
// dense GEMMs hs = relu(xs@Wg^T+bg) and hs_p = relu(xs_p@Wa^T+ba), with the
// bf16 weight tile staged into LDS via the Tensor Data Mover (tensor_load_to_lds,
// TENSORcnt) so B-fragments come from LDS instead of per-wave global re-reads.
// All reductions are fixed-order (no float atomics) => deterministic output.

typedef __attribute__((ext_vector_type(16))) __bf16 bf16x16;
typedef __attribute__((ext_vector_type(8)))  __bf16 bf16x8;
typedef __attribute__((ext_vector_type(8)))  float  floatx8;
typedef __attribute__((ext_vector_type(4)))  unsigned int v4u;
typedef __attribute__((ext_vector_type(8)))  int  v8i;
typedef __attribute__((ext_vector_type(4)))  int  v4i;
typedef unsigned short u16;

#define NA   16384   // atoms
#define NW   16384   // protein words
#define DIMV 128
#define CAP  128     // max stored nonzeros/row (mean 32.8, std 5.7 -> 16.6 sigma)

// ---------------- embedding gather: out[i][d] = table[idx[i]][d] ----------------
__global__ __launch_bounds__(128) void k_gather(const int* __restrict__ idx,
                                                const float* __restrict__ table,
                                                float* __restrict__ out) {
  int i = blockIdx.x, d = threadIdx.x;
  out[(size_t)i * DIMV + d] = table[(size_t)idx[i] * DIMV + d];
}

// ---------------- fp32 -> bf16 weight pre-conversion ----------------
__global__ __launch_bounds__(256) void k_f32_to_bf16(const float* __restrict__ src,
                                                     __bf16* __restrict__ dst, int n) {
  int i = blockIdx.x * 256 + threadIdx.x;
  if (i < n) dst[i] = (__bf16)src[i];
}

// ---------------- sparsify adjacency: one wave per row, ordered compaction -------
// Deterministic: columns stored in ascending order via ballot/popcount prefix.
__global__ __launch_bounds__(256) void k_sparsify(const float* __restrict__ A,
                                                  u16* __restrict__ cols,
                                                  float* __restrict__ vals,
                                                  int* __restrict__ nnz) {
  const int lane = threadIdx.x & 31;
  const int wave = threadIdx.x >> 5;
  const int row  = blockIdx.x * 8 + wave;
  const float* Ar = A + (size_t)row * NA;
  u16*   c = cols + (size_t)row * CAP;
  float* v = vals + (size_t)row * CAP;
  int base = 0;
  for (int k0 = 0; k0 < NA; k0 += 32) {
    float a = Ar[k0 + lane];
    __builtin_prefetch(Ar + k0 + 2048, 0, 0);
    unsigned mask = (unsigned)__ballot(a != 0.f);      // wave32: low 32 bits
    if (a != 0.f) {
      int off = base + __popc(mask & ((1u << lane) - 1u));
      if (off < CAP) { c[off] = (u16)(k0 + lane); v[off] = a; }
    }
    base += __popc(mask);
  }
  if (lane == 0) nnz[row] = base;   // raw count; >CAP triggers dense fallback
}

// ---------------- sparse xs += A @ hs (gather-sum from L2-resident hs) -----------
__global__ __launch_bounds__(128) void k_sparse_spmm(const float* __restrict__ A,
                                                     const u16* __restrict__ cols,
                                                     const float* __restrict__ vals,
                                                     const int* __restrict__ nnz,
                                                     const float* __restrict__ hs,
                                                     float* __restrict__ xs) {
  __shared__ u16   scol[CAP];
  __shared__ float sval[CAP];
  const int i = blockIdx.x, d = threadIdx.x;
  float acc = xs[(size_t)i * DIMV + d];
  const int n = nnz[i];
  if (n <= CAP) {
    if (d < n) { scol[d] = cols[(size_t)i * CAP + d]; sval[d] = vals[(size_t)i * CAP + d]; }
    __syncthreads();
    for (int j = 0; j < n; ++j)
      acc = fmaf(sval[j], hs[(size_t)scol[j] * DIMV + d], acc);
  } else {
    // astronomically rare overflow: exact dense fallback, fixed order
    const float* Ar = A + (size_t)i * NA;
    for (int k = 0; k < NA; ++k) {
      float a = Ar[k];
      if (a != 0.f) acc = fmaf(a, hs[(size_t)k * DIMV + d], acc);
    }
  }
  xs[(size_t)i * DIMV + d] = acc;
}

// ---------------- WMMA dense: out = relu(X @ W^T + b), X:[M,128] fp32, W bf16 ----
// Block: 128 thr = 4 waves; M-tile 64 (wave w -> 16-row subtile), N = 128 (8 subtiles).
// The 128x128 bf16 weight tile (32KB) is staged into LDS once per workgroup by the
// Tensor Data Mover (2D D# descriptor per CDNA5 ISA 8.3/8.4), then B-fragments are
// contiguous 16B ds_load_b128 reads.
// bf16 16x16x32 fragment layout:
//   lanes 0-15 : row = lane,    VGPR0-3 = K 0..7,  VGPR4-7 = K 16..23
//   lanes 16-31: row = lane-16, VGPR0-3 = K 8..15, VGPR4-7 = K 24..31
__global__ __launch_bounds__(128) void k_wmma_dense_relu(const float* __restrict__ X,
                                                         const __bf16* __restrict__ Wb,
                                                         const float* __restrict__ bias,
                                                         float* __restrict__ out) {
  __shared__ __bf16 sW[DIMV * DIMV];   // 32 KB

#if defined(__gfx1250__) && __has_builtin(__builtin_amdgcn_tensor_load_to_lds)
  if (threadIdx.x == 0) {
    // D# group0: count=1 | lds_addr | 57b global_addr | type=2 ("image")
    unsigned lds_off = (unsigned)(size_t)(void*)sW;       // low 32 bits = LDS offset
    unsigned long long ga = (unsigned long long)(size_t)Wb;
    v4u g0;
    g0[0] = 1u;                                           // count=1 (valid user D#)
    g0[1] = lds_off;                                      // lds_addr (bytes)
    g0[2] = (unsigned)(ga & 0xffffffffu);                 // global_addr[31:0]
    g0[3] = (unsigned)((ga >> 32) & 0x1ffffffu) | (2u << 30); // [56:32] | type=2
    // D# group1: data_size=2B; tensor 128x128; tile 128x128; dim0_stride=128
    v8i g1;
    g1[0] = (1 << 16);          // workgroup_mask=0, data_size=1 (2 bytes)
    g1[1] = (128 << 16);        // tensor_dim0[15:0]=128 (bits 63:48)
    g1[2] = (128 << 16);        // tensor_dim0[31:16]=0 | tensor_dim1[15:0]=128
    g1[3] = (128 << 16);        // tensor_dim1[31:16]=0 | tile_dim0=128
    g1[4] = 128;                // tile_dim1=128, tile_dim2=0
    g1[5] = 128;                // tensor_dim0_stride[31:0]=128
    g1[6] = 0;                  // stride0 hi | stride1 lo
    g1[7] = 0;
    v4i g2 = {0, 0, 0, 0}, g3 = {0, 0, 0, 0};             // unused (2D tensor)
    v8i g4 = {0, 0, 0, 0, 0, 0, 0, 0};                    // unused trailing group
    // amdgpu-toolchain (clang-23) 6-arg form
    __builtin_amdgcn_tensor_load_to_lds(g0, g1, g2, g3, g4, 0);
    __builtin_amdgcn_s_wait_tensorcnt(0);
  }
#else
  for (int e = threadIdx.x; e < (DIMV * DIMV) / 8; e += 128)
    ((uint4*)sW)[e] = ((const uint4*)Wb)[e];
#endif
  __syncthreads();

  const int lane = threadIdx.x & 31;
  const int wave = threadIdx.x >> 5;
  const int l15  = lane & 15;
  const int hi   = lane >> 4;
  const int mrow0 = blockIdx.x * 64 + wave * 16;

  floatx8 acc[8];
#pragma unroll
  for (int nt = 0; nt < 8; ++nt) {
    float b = bias[nt * 16 + l15];
#pragma unroll
    for (int r = 0; r < 8; ++r) acc[nt][r] = b;
  }

  const float* Xr = X + (size_t)(mrow0 + l15) * DIMV;

#pragma unroll
  for (int kb = 0; kb < DIMV; kb += 32) {
    const int ka = kb + hi * 8;
    float4 a0 = *(const float4*)(Xr + ka);
    float4 a1 = *(const float4*)(Xr + ka + 4);
    float4 a2 = *(const float4*)(Xr + ka + 16);
    float4 a3 = *(const float4*)(Xr + ka + 20);
    bf16x16 av;
    av[0]  = (__bf16)a0.x; av[1]  = (__bf16)a0.y; av[2]  = (__bf16)a0.z; av[3]  = (__bf16)a0.w;
    av[4]  = (__bf16)a1.x; av[5]  = (__bf16)a1.y; av[6]  = (__bf16)a1.z; av[7]  = (__bf16)a1.w;
    av[8]  = (__bf16)a2.x; av[9]  = (__bf16)a2.y; av[10] = (__bf16)a2.z; av[11] = (__bf16)a2.w;
    av[12] = (__bf16)a3.x; av[13] = (__bf16)a3.y; av[14] = (__bf16)a3.z; av[15] = (__bf16)a3.w;

#pragma unroll
    for (int nt = 0; nt < 8; ++nt) {
      const __bf16* bp = sW + (size_t)(nt * 16 + l15) * DIMV + ka;
      bf16x8 b0 = *(const bf16x8*)(bp);
      bf16x8 b1 = *(const bf16x8*)(bp + 16);
      bf16x16 bv = __builtin_shufflevector(b0, b1, 0, 1, 2, 3, 4, 5, 6, 7,
                                           8, 9, 10, 11, 12, 13, 14, 15);
      acc[nt] = __builtin_amdgcn_wmma_f32_16x16x32_bf16(
          false, av, false, bv, (short)0, acc[nt], false, false);
    }
  }

#pragma unroll
  for (int nt = 0; nt < 8; ++nt) {
#pragma unroll
    for (int r = 0; r < 8; ++r) {
      int m = mrow0 + r + hi * 8;
      out[(size_t)m * DIMV + nt * 16 + l15] = fmaxf(acc[nt][r], 0.f);
    }
  }
}

// ---------------- 11x11 same-padded conv + bias + relu, LDS-tiled ----------------
#define CROWS 2
__global__ __launch_bounds__(256) void k_conv11(const float* __restrict__ in,
                                                float* __restrict__ out,
                                                const float* __restrict__ kk,
                                                const float* __restrict__ kb) {
  __shared__ float tile[CROWS + 10][140];   // (rows+10) x (128+10), padded stride
  const int t  = threadIdx.x;
  const int y0 = blockIdx.x * CROWS;
  for (int e = t; e < (CROWS + 10) * 138; e += 256) {
    int r = e / 138, c = e - r * 138;
    int yy = y0 - 5 + r, xx = c - 5;
    float v = 0.f;
    if (yy >= 0 && yy < NW && xx >= 0 && xx < DIMV) v = in[(size_t)yy * DIMV + xx];
    tile[r][c] = v;
  }
  __syncthreads();
  const int ry = t >> 7, x = t & 127;
  float s = kb[0];
#pragma unroll
  for (int dy = 0; dy < 11; ++dy)
#pragma unroll
    for (int dx = 0; dx < 11; ++dx)
      s = fmaf(tile[ry + dy][x + dx], kk[dy * 11 + dx], s);
  out[(size_t)(y0 + ry) * DIMV + x] = fmaxf(s, 0.f);
}

// ---------------- deterministic two-pass (weighted) column mean ----------------
__global__ __launch_bounds__(128) void k_colmean_part(const float* __restrict__ m,
                                                      const float* __restrict__ w, // nullable
                                                      float* __restrict__ partial, // [64][128]
                                                      int rows) {
  int d = threadIdx.x;
  float s = 0.f;
  for (int r = blockIdx.x; r < rows; r += 64) {
    float v = m[(size_t)r * DIMV + d];
    s += w ? w[r] * v : v;
  }
  partial[(size_t)blockIdx.x * DIMV + d] = s;
}

__global__ __launch_bounds__(128) void k_colmean_fin(const float* __restrict__ partial,
                                                     float* __restrict__ out, float scale) {
  int d = threadIdx.x;
  float s = 0.f;
  for (int b = 0; b < 64; ++b) s += partial[(size_t)b * DIMV + d];
  out[d] = s * scale;
}

// ---------------- h = relu(ba + Wa @ compound) ----------------
__global__ __launch_bounds__(128) void k_small_mv(const float* __restrict__ v,
                                                  const float* __restrict__ W,
                                                  const float* __restrict__ b,
                                                  float* __restrict__ out) {
  __shared__ float row[DIMV];
  int j = threadIdx.x;
  row[j] = v[j];
  __syncthreads();
  float s = b[j];
  const float* w = W + (size_t)j * DIMV;
#pragma unroll 8
  for (int d = 0; d < DIMV; ++d) s = fmaf(row[d], w[d], s);
  out[j] = fmaxf(s, 0.f);
}

// ---------------- weights[l] = tanh(dot(h, hs_p[l])) ----------------
__global__ __launch_bounds__(128) void k_attn_weights(const float* __restrict__ hsp,
                                                      const float* __restrict__ h,
                                                      float* __restrict__ wout) {
  __shared__ float red[DIMV];
  int l = blockIdx.x, t = threadIdx.x;
  red[t] = hsp[(size_t)l * DIMV + t] * h[t];
  __syncthreads();
  for (int s = 64; s > 0; s >>= 1) {
    if (t < s) red[t] += red[t + s];
    __syncthreads();
  }
  if (t == 0) wout[l] = tanhf(red[0]);
}

// ---------------- output MLP ----------------
__global__ __launch_bounds__(256) void k_mlp(const float* __restrict__ compound,
                                             const float* __restrict__ protein,
                                             const float* __restrict__ Wo,
                                             const float* __restrict__ bo,
                                             const float* __restrict__ Wi,
                                             const float* __restrict__ bi,
                                             float* __restrict__ out) {
  __shared__ float cat[256], nxt[256];
  int t = threadIdx.x;
  cat[t] = (t < 128) ? compound[t] : protein[t - 128];
  __syncthreads();
  for (int L = 0; L < 2; ++L) {
    float s = bo[t];
    const float* w = Wo + (size_t)t * 256;
#pragma unroll 8
    for (int k = 0; k < 256; ++k) s = fmaf(cat[k], w[k], s);
    nxt[t] = fmaxf(s, 0.f);
    __syncthreads();
    cat[t] = nxt[t];
    __syncthreads();
  }
  if (t < 2) {
    float s = bi[t];
    const float* w = Wi + (size_t)t * 256;
    for (int k = 0; k < 256; ++k) s = fmaf(cat[k], w[k], s);
    out[t] = s;
  }
}

extern "C" void kernel_launch(void* const* d_in, const int* in_sizes, int n_in,
                              void* d_out, int out_size, void* d_ws, size_t ws_size,
                              hipStream_t stream) {
  const int*   fingerprints = (const int*)  d_in[0];
  const float* adjacency    = (const float*)d_in[1];
  const int*   words        = (const int*)  d_in[2];
  const float* embed_fp     = (const float*)d_in[3];
  const float* embed_word   = (const float*)d_in[4];
  const float* Wg           = (const float*)d_in[5];
  const float* bg           = (const float*)d_in[6];
  const float* conv_k       = (const float*)d_in[7];
  const float* conv_b       = (const float*)d_in[8];
  const float* Wa           = (const float*)d_in[9];
  const float* ba           = (const float*)d_in[10];
  const float* Wo           = (const float*)d_in[11];
  const float* bo           = (const float*)d_in[12];
  const float* Wi           = (const float*)d_in[13];
  const float* bi           = (const float*)d_in[14];
  float* out = (float*)d_out;

  // workspace carve-up (~44.3 MB)
  char* ws = (char*)d_ws;
  float*  xs       = (float*) (ws);                             // 8 MB
  float*  hs       = (float*) (ws + (8u  << 20));               // 8 MB
  float*  P0       = (float*) (ws + (16u << 20));               // 8 MB
  float*  P1       = (float*) (ws + (24u << 20));               // 8 MB
  u16*    cols     = (u16*)   (ws + (32u << 20));               // 4 MB
  float*  vals     = (float*) (ws + (36u << 20));               // 8 MB
  int*    nnz      = (int*)   (ws + (44u << 20));               // 64 KB
  __bf16* Wgb      = (__bf16*)(ws + (44u << 20) + (64u << 10)); // 32 KB
  __bf16* Wab      = (__bf16*)(ws + (44u << 20) + (96u << 10)); // 32 KB
  float*  weights  = (float*) (ws + (44u << 20) + (128u << 10));// 64 KB
  float*  partial  = (float*) (ws + (44u << 20) + (192u << 10));// 32 KB
  float*  compound = (float*) (ws + (44u << 20) + (224u << 10));
  float*  protein  = compound + 128;
  float*  hvec     = compound + 256;

  // ---- one-time prep: bf16 weights + adjacency sparsification ----
  k_f32_to_bf16<<<(DIMV * DIMV + 255) / 256, 256, 0, stream>>>(Wg, Wgb, DIMV * DIMV);
  k_f32_to_bf16<<<(DIMV * DIMV + 255) / 256, 256, 0, stream>>>(Wa, Wab, DIMV * DIMV);
  k_sparsify<<<NA / 8, 256, 0, stream>>>(adjacency, cols, vals, nnz);

  // ---- GNN: xs = embed_fp[fingerprints]; 3x (WMMA hs; sparse xs += A hs) ----
  k_gather<<<NA, DIMV, 0, stream>>>(fingerprints, embed_fp, xs);
  for (int l = 0; l < 3; ++l) {
    k_wmma_dense_relu<<<NA / 64, 128, 0, stream>>>(xs, Wgb, bg, hs);
    k_sparse_spmm<<<NA, DIMV, 0, stream>>>(adjacency, cols, vals, nnz, hs, xs);
  }
  k_colmean_part<<<64, DIMV, 0, stream>>>(xs, nullptr, partial, NA);
  k_colmean_fin<<<1, DIMV, 0, stream>>>(partial, compound, 1.0f / NA);

  // ---- CNN over protein words: 3x 11x11 conv+relu (LDS-tiled, ping-pong) ----
  k_gather<<<NW, DIMV, 0, stream>>>(words, embed_word, P0);
  k_conv11<<<NW / CROWS, 256, 0, stream>>>(P0, P1, conv_k, conv_b);
  k_conv11<<<NW / CROWS, 256, 0, stream>>>(P1, P0, conv_k, conv_b);
  k_conv11<<<NW / CROWS, 256, 0, stream>>>(P0, P1, conv_k, conv_b);   // xs_p in P1

  // ---- attention pooling ----
  k_wmma_dense_relu<<<NW / 64, 128, 0, stream>>>(P1, Wab, ba, P0);    // hs_p in P0
  k_small_mv<<<1, DIMV, 0, stream>>>(compound, Wa, ba, hvec);         // h
  k_attn_weights<<<NW, DIMV, 0, stream>>>(P0, hvec, weights);         // tanh(h . hs_p[l])
  k_colmean_part<<<64, DIMV, 0, stream>>>(P0, weights, partial, NW);
  k_colmean_fin<<<1, DIMV, 0, stream>>>(partial, protein, 1.0f / NW);

  // ---- output MLP ----
  k_mlp<<<1, 256, 0, stream>>>(compound, protein, Wo, bo, Wi, bi, out);
}
